// LSTM_30142080483558
// MI455X (gfx1250) — matrix operand order repaired and checked by reading, not compiled, not run on hardware
//
#include <hip/hip_runtime.h>
#include <hip/hip_bf16.h>
#include <cstdint>
#include <cstddef>

#define H_DIM 1024
#define V_DIM 4096
#define B_DIM 64
#define T_DIM 256

typedef __attribute__((ext_vector_type(16))) __bf16        v16bf;
typedef __attribute__((ext_vector_type(8)))  float         v8f;
typedef __attribute__((ext_vector_type(4)))  unsigned int  u32x4;

union FragBF16 { u32x4 q[2]; v16bf m; };

__device__ __forceinline__ unsigned short f32_to_bf16_rne(float f) {
    unsigned int u = __float_as_uint(f);
    u += 0x7FFFu + ((u >> 16) & 1u);
    return (unsigned short)(u >> 16);
}

__device__ __forceinline__ float sigmoidf(float x) {
    return 1.0f / (1.0f + expf(-x));
}

// Stage `nchunks` 16-byte chunks from global `src` into LDS at `lds_base`
// using the CDNA5 async global->LDS DMA path (ASYNCcnt).
//
// The LDS address operand is the low 32 bits of the generic pointer: for the
// LDS aperture, flat addresses carry the wave-relative logical LDS byte
// address in bits [31:0] (ISA 10.2 aperture rules), which is exactly what the
// async instruction's VDST operand expects (hardware adds LDS_BASE).
// Critically, the ptrtoint of the shared pointer also *captures* the LDS
// object, so with the "memory" clobber the compiler must preserve subsequent
// ds_load reads of data written by the DMA engine behind this asm.
__device__ __forceinline__ void async_stage_to_lds(unsigned short* lds_base,
                                                   const unsigned short* src,
                                                   int nchunks, int tid, int nthreads) {
    unsigned int lds_addr0 = (unsigned int)(size_t)(void*)lds_base;
    for (int i = tid; i < nchunks; i += nthreads) {
        unsigned int lds_off = lds_addr0 + (unsigned int)(i * 16);
        const unsigned short* g = src + i * 8;
        asm volatile("global_load_async_to_lds_b128 %0, %1, off"
                     :: "v"(lds_off), "v"(g) : "memory");
    }
    asm volatile("s_wait_asynccnt 0" ::: "memory");
}

// ---------------------------------------------------------------------------
// Kernel 0: convert fp32 weights -> bf16, pre-transposed to (N, K) K-contiguous
//   whh_t: (4096, 1024)  rows = [f | i | c | o] output columns
//   yw_t : (4096, 1024)  rows = vocab columns of y_w_ho
// ---------------------------------------------------------------------------
__global__ void convert_weights(const float* __restrict__ f_hh,
                                const float* __restrict__ i_hh,
                                const float* __restrict__ c_hh,
                                const float* __restrict__ o_hh,
                                const float* __restrict__ y_w,       // (H, V)
                                unsigned short* __restrict__ whh_t,  // (4096,1024)
                                unsigned short* __restrict__ yw_t) {
    int idx = blockIdx.x * blockDim.x + threadIdx.x;   // 0 .. 4096*1024-1
    int n = idx >> 10;          // 0..4095
    int k = idx & 1023;         // 0..1023
    int gate = n >> 10;         // 0..3
    int h = n & 1023;
    const float* g = (gate == 0) ? f_hh : (gate == 1) ? i_hh : (gate == 2) ? c_hh : o_hh;
    whh_t[idx] = f32_to_bf16_rne(g[(size_t)k * H_DIM + h]);
    yw_t[idx]  = f32_to_bf16_rne(y_w[(size_t)k * V_DIM + n]);
}

// ---------------------------------------------------------------------------
// Kernel 1: initialize recurrent state from H0 / C0
// ---------------------------------------------------------------------------
__global__ void init_state(const float* __restrict__ H0, const float* __restrict__ C0,
                           unsigned short* __restrict__ Hb, float* __restrict__ C) {
    int idx = blockIdx.x * blockDim.x + threadIdx.x;   // 0 .. 64*1024-1
    Hb[idx] = f32_to_bf16_rne(H0[idx]);
    C[idx]  = C0[idx];
}

// ---------------------------------------------------------------------------
// Kernel 2: per-step gate GEMM:  gates(64,4096) = H(64,1024)bf16 @ Whh_t^T
//   block = 128 threads (4 waves). Each wave: one 16x16 tile, K=1024.
//   A tile (16 x 1024 bf16, 32 KB) async-DMA'd into LDS, shared by the 4 waves.
//   grid = (64, 4): blockIdx.x -> N tile group of 64, blockIdx.y -> M tile.
// ---------------------------------------------------------------------------
__global__ void gate_gemm(const unsigned short* __restrict__ Hb,     // (64,1024)
                          const unsigned short* __restrict__ whh_t,  // (4096,1024)
                          float* __restrict__ gates) {               // (64,4096)
    __shared__ __attribute__((aligned(16))) unsigned short ldsA[16 * H_DIM];

    const int tid  = threadIdx.x;
    const int wave = tid >> 5;
    const int lane = tid & 31;
    const int half = lane >> 4;
    const int l16  = lane & 15;
    const int m0   = blockIdx.y << 4;
    const int n    = (blockIdx.x << 6) + (wave << 4) + l16;

    // Async-stage the contiguous 32KB A tile (rows m0..m0+15) into LDS.
    async_stage_to_lds(ldsA, Hb + (size_t)m0 * H_DIM, (16 * H_DIM * 2) / 16, tid, 128);
    __syncthreads();

    const unsigned short* aptr = ldsA + l16 * H_DIM + half * 16;
    const unsigned short* bptr = whh_t + (size_t)n * H_DIM + half * 16;

    v8f acc = {};
    for (int k0 = 0; k0 < H_DIM; k0 += 32) {
        FragBF16 a, b;
        a.q[0] = *(const u32x4*)(aptr + k0);
        a.q[1] = *(const u32x4*)(aptr + k0 + 8);
        b.q[0] = *(const u32x4*)(bptr + k0);
        b.q[1] = *(const u32x4*)(bptr + k0 + 8);
        acc = __builtin_amdgcn_wmma_f32_16x16x32_bf16(false, a.m, false, b.m,
                                                      (short)0, acc, false, false);
    }

#pragma unroll
    for (int j = 0; j < 8; ++j) {
        int m = m0 + j + half * 8;
        gates[(size_t)m * V_DIM + n] = acc[j];
    }
}

// ---------------------------------------------------------------------------
// Kernel 3: cell update (elementwise over (m,h)):
//   gate = act(raw + w_xh[token] + bias); C' = f*C + i*c~ ; H' = o*C'
//   writes H' as bf16 (next-step state + history) and fp32 Hf/Cf at t==T-1.
// ---------------------------------------------------------------------------
__global__ void cell_update(const float* __restrict__ gates,    // (64,4096) raw
                            const int*   __restrict__ tokens,   // (B, T)
                            const float* __restrict__ fx, const float* __restrict__ fb,
                            const float* __restrict__ ix, const float* __restrict__ ib,
                            const float* __restrict__ cx, const float* __restrict__ cb,
                            const float* __restrict__ ox, const float* __restrict__ ob,
                            float* __restrict__ C,
                            unsigned short* __restrict__ Hb,
                            unsigned short* __restrict__ Hhist,
                            float* __restrict__ HfCf,           // [Hf(64,1024) | Cf(64,1024)]
                            int t) {
    int idx = blockIdx.x * blockDim.x + threadIdx.x;  // 0 .. 64*1024-1
    int m = idx >> 10;
    int h = idx & 1023;
    int tok = tokens[m * T_DIM + t];
    size_t xo = (size_t)tok * H_DIM + h;
    const float* g = gates + (size_t)m * V_DIM;

    float fg = sigmoidf(g[h]              + fx[xo] + fb[h]);
    float ig = sigmoidf(g[H_DIM + h]      + ix[xo] + ib[h]);
    float cg = tanhf   (g[2 * H_DIM + h]  + cx[xo] + cb[h]);
    float og = sigmoidf(g[3 * H_DIM + h]  + ox[xo] + ob[h]);

    float cn = fg * C[idx] + ig * cg;
    float hn = og * cn;                  // faithful: no tanh on cell output

    C[idx] = cn;
    unsigned short hb = f32_to_bf16_rne(hn);
    Hb[idx] = hb;
    Hhist[(size_t)t * (B_DIM * H_DIM) + idx] = hb;
    if (t == T_DIM - 1) {
        HfCf[idx] = hn;
        HfCf[B_DIM * H_DIM + idx] = cn;
    }
}

// ---------------------------------------------------------------------------
// Kernel 4: batched output projection  Y(16384,4096) = Hhist(16384,1024) @ yw^T + b
//   block = 128 threads (4 waves); each wave computes 16x64 (4 N-subtiles)
//   reusing one A fragment per K step. A tile async-DMA'd into LDS.
//   grid = (4096/256, 16384/16) = (16, 1024)
// ---------------------------------------------------------------------------
__global__ void out_gemm(const unsigned short* __restrict__ Hhist, // (16384,1024)
                         const unsigned short* __restrict__ yw_t,  // (4096,1024)
                         const float* __restrict__ yb,             // (4096)
                         float* __restrict__ Y) {                  // (16384,4096)
    __shared__ __attribute__((aligned(16))) unsigned short ldsA[16 * H_DIM];

    const int tid  = threadIdx.x;
    const int wave = tid >> 5;
    const int lane = tid & 31;
    const int half = lane >> 4;
    const int l16  = lane & 15;
    const int m0   = blockIdx.y << 4;
    const int nw   = (blockIdx.x << 8) + (wave << 6);   // wave's 64-col base

    async_stage_to_lds(ldsA, Hhist + (size_t)m0 * H_DIM, (16 * H_DIM * 2) / 16, tid, 128);
    __syncthreads();

    const unsigned short* aptr = ldsA + l16 * H_DIM + half * 16;
    const unsigned short* b0 = yw_t + (size_t)(nw +  0 + l16) * H_DIM + half * 16;
    const unsigned short* b1 = yw_t + (size_t)(nw + 16 + l16) * H_DIM + half * 16;
    const unsigned short* b2 = yw_t + (size_t)(nw + 32 + l16) * H_DIM + half * 16;
    const unsigned short* b3 = yw_t + (size_t)(nw + 48 + l16) * H_DIM + half * 16;

    v8f acc0 = {}, acc1 = {}, acc2 = {}, acc3 = {};
    for (int k0 = 0; k0 < H_DIM; k0 += 32) {
        FragBF16 a;
        a.q[0] = *(const u32x4*)(aptr + k0);
        a.q[1] = *(const u32x4*)(aptr + k0 + 8);

        __builtin_prefetch(b0 + k0 + 128, 0, 1);   // global_prefetch for streaming B

        FragBF16 b;
        b.q[0] = *(const u32x4*)(b0 + k0); b.q[1] = *(const u32x4*)(b0 + k0 + 8);
        acc0 = __builtin_amdgcn_wmma_f32_16x16x32_bf16(false, a.m, false, b.m,
                                                       (short)0, acc0, false, false);
        b.q[0] = *(const u32x4*)(b1 + k0); b.q[1] = *(const u32x4*)(b1 + k0 + 8);
        acc1 = __builtin_amdgcn_wmma_f32_16x16x32_bf16(false, a.m, false, b.m,
                                                       (short)0, acc1, false, false);
        b.q[0] = *(const u32x4*)(b2 + k0); b.q[1] = *(const u32x4*)(b2 + k0 + 8);
        acc2 = __builtin_amdgcn_wmma_f32_16x16x32_bf16(false, a.m, false, b.m,
                                                       (short)0, acc2, false, false);
        b.q[0] = *(const u32x4*)(b3 + k0); b.q[1] = *(const u32x4*)(b3 + k0 + 8);
        acc3 = __builtin_amdgcn_wmma_f32_16x16x32_bf16(false, a.m, false, b.m,
                                                       (short)0, acc3, false, false);
    }

#pragma unroll
    for (int s = 0; s < 4; ++s) {
        v8f acc = (s == 0) ? acc0 : (s == 1) ? acc1 : (s == 2) ? acc2 : acc3;
        int n = nw + s * 16 + l16;
        float bias = yb[n];
#pragma unroll
        for (int j = 0; j < 8; ++j) {
            int m = m0 + j + half * 8;
            Y[(size_t)m * V_DIM + n] = acc[j] + bias;
        }
    }
}

// ---------------------------------------------------------------------------
extern "C" void kernel_launch(void* const* d_in, const int* in_sizes, int n_in,
                              void* d_out, int out_size, void* d_ws, size_t ws_size,
                              hipStream_t stream) {
    (void)in_sizes; (void)n_in; (void)out_size; (void)ws_size;

    const int*   tokens = (const int*)  d_in[0];   // (B, T)
    const float* H0     = (const float*)d_in[1];
    const float* C0     = (const float*)d_in[2];
    const float* f_w_xh = (const float*)d_in[3];
    const float* f_w_hh = (const float*)d_in[4];
    const float* f_b    = (const float*)d_in[5];
    const float* i_w_xh = (const float*)d_in[6];
    const float* i_w_hh = (const float*)d_in[7];
    const float* i_b    = (const float*)d_in[8];
    const float* c_w_xh = (const float*)d_in[9];
    const float* c_w_hh = (const float*)d_in[10];
    const float* c_b    = (const float*)d_in[11];
    const float* o_w_xh = (const float*)d_in[12];
    const float* o_w_hh = (const float*)d_in[13];
    const float* o_b    = (const float*)d_in[14];
    const float* y_w_ho = (const float*)d_in[15];
    const float* y_b_o  = (const float*)d_in[16];

    // Workspace layout (bytes):
    char* ws = (char*)d_ws;
    unsigned short* whh_t = (unsigned short*)(ws);                         // 8 MiB
    unsigned short* yw_t  = (unsigned short*)(ws + (8u  << 20));           // 8 MiB
    unsigned short* Hhist = (unsigned short*)(ws + (16u << 20));           // 32 MiB
    unsigned short* Hb    = (unsigned short*)(ws + (48u << 20));           // 128 KiB
    float*          Cst   = (float*)        (ws + (48u << 20) + 131072);   // 256 KiB
    float*          gates = (float*)        (ws + (48u << 20) + 393216);   // 1 MiB

    float* Y    = (float*)d_out;                                   // (T,B,V)
    float* HfCf = Y + (size_t)T_DIM * B_DIM * V_DIM;               // Hf | Cf

    convert_weights<<<(4096 * 1024) / 256, 256, 0, stream>>>(
        f_w_hh, i_w_hh, c_w_hh, o_w_hh, y_w_ho, whh_t, yw_t);
    init_state<<<(B_DIM * H_DIM) / 256, 256, 0, stream>>>(H0, C0, Hb, Cst);

    for (int t = 0; t < T_DIM; ++t) {
        gate_gemm<<<dim3(V_DIM / 64, B_DIM / 16), 128, 0, stream>>>(Hb, whh_t, gates);
        cell_update<<<(B_DIM * H_DIM) / 256, 256, 0, stream>>>(
            gates, tokens,
            f_w_xh, f_b, i_w_xh, i_b, c_w_xh, c_b, o_w_xh, o_b,
            Cst, Hb, Hhist, HfCf, t);
    }

    out_gemm<<<dim3(V_DIM / 256, (T_DIM * B_DIM) / 16), 128, 0, stream>>>(
        Hhist, yw_t, y_b_o, Y);
}
